// LstmReluGraphSage_38732015076059
// MI455X (gfx1250) — compile-verified, matrix operand order checked by reading, not run on hardware
//
#include <hip/hip_runtime.h>
#include <hip/hip_bf16.h>
#include <hip/hip_fp16.h>

#define NN    50000
#define DD    16
#define EE    (NN*DD)
#define INN   128
#define INE   64
#define HH    64
#define HH2   128
#define G4    512
#define OUTN  128
#define OUTE  64

typedef __attribute__((ext_vector_type(16))) _Float16 v16h;
typedef __attribute__((ext_vector_type(8)))  float    v8f;

// ---------------------------------------------------------------- helpers

__device__ __forceinline__ v8f wmma32(v16h a, v16h b, v8f c) {
  // D = A(16x32 f16) * B(32x16 f16) + C(16x16 f32)
  return __builtin_amdgcn_wmma_f32_16x16x32_f16(false, a, false, b, (short)0, c,
                                                false, false);
}

__device__ __forceinline__ v16h mk16(uint4 u0, uint4 u1) {
  union U { uint4 u[2]; v16h v; } r;
  r.u[0] = u0; r.u[1] = u1;
  return r.v;
}

// A/B chunk load from f16 row (row-major, K contiguous), k-chunk kc, lane half kb0.
// A-layout (16-bit, 16x32): elems 0..7 -> K = kc*32+kb0 .. +7 ; elems 8..15 -> +16.
__device__ __forceinline__ v16h ld16(const _Float16* row, int kc, int kb0) {
  const uint4* p0 = (const uint4*)(row + kc * 32 + kb0);
  const uint4* p1 = (const uint4*)(row + kc * 32 + 16 + kb0);
  return mk16(*p0, *p1);
}

// f32 row load with on-the-fly fp16 conversion (same layout).
__device__ __forceinline__ v16h ld32cvt(const float* row, int kc, int kb0) {
  const float4* p0 = (const float4*)(row + kc * 32 + kb0);
  const float4* p1 = (const float4*)(row + kc * 32 + 16 + kb0);
  float4 a = p0[0], b = p0[1], c = p1[0], d = p1[1];
  v16h v;
  v[0]  = (_Float16)a.x; v[1]  = (_Float16)a.y; v[2]  = (_Float16)a.z; v[3]  = (_Float16)a.w;
  v[4]  = (_Float16)b.x; v[5]  = (_Float16)b.y; v[6]  = (_Float16)b.z; v[7]  = (_Float16)b.w;
  v[8]  = (_Float16)c.x; v[9]  = (_Float16)c.y; v[10] = (_Float16)c.z; v[11] = (_Float16)c.w;
  v[12] = (_Float16)d.x; v[13] = (_Float16)d.y; v[14] = (_Float16)d.z; v[15] = (_Float16)d.w;
  return v;
}

// select segment pointer for concatenated [seg0(64) | seg1(64) | seg2(64)] rows
__device__ __forceinline__ const _Float16* segp(const _Float16* s0,
                                                const _Float16* s1,
                                                const _Float16* s2, int k) {
  return k < 64 ? (s0 + k) : (k < 128 ? (s1 + (k - 64)) : (s2 + (k - 128)));
}

__device__ __forceinline__ float sigm(float x) { return 1.0f / (1.0f + __expf(-x)); }
__device__ __forceinline__ float tanh_f(float x) {
  x = fminf(fmaxf(x, -15.0f), 15.0f);
  float e = __expf(2.0f * x);
  return (e - 1.0f) / (e + 1.0f);
}

// ---------------------------------------------------------------- small utils

__global__ __launch_bounds__(256) void cvt_f32_to_f16(const float* __restrict__ s,
                                                      _Float16* __restrict__ d, int n) {
  int i = blockIdx.x * blockDim.x + threadIdx.x;
  if (i < n) d[i] = (_Float16)s[i];
}

__global__ __launch_bounds__(256) void add_bias(const float* __restrict__ a,
                                                const float* __restrict__ b,
                                                float* __restrict__ o, int n) {
  int i = blockIdx.x * blockDim.x + threadIdx.x;
  if (i < n) o[i] = a[i] + b[i];
}

// ---------------------------------------------------------------- pretrans GEMM
// out_h[M, NT*16] = relu( A[M, KC*32] @ W[NT*16, KC*32]^T + bias ), stored fp16.
template <int KC, int NT>
__global__ __launch_bounds__(256)
void pretrans_gemm(const float* __restrict__ A, const _Float16* __restrict__ W,
                   const float* __restrict__ bias, _Float16* __restrict__ outh,
                   int mtiles) {
  int gw = (int)((blockIdx.x * blockDim.x + threadIdx.x) >> 5);
  if (gw >= mtiles) return;
  int lane = threadIdx.x & 31;
  int l15 = lane & 15, hi = lane >> 4, kb0 = hi * 8;
  int mbase = gw * 16;
  const int K = KC * 32;
  const float* arow = A + (size_t)(mbase + l15) * K;
  v16h a[KC];
#pragma unroll
  for (int kc = 0; kc < KC; ++kc) a[kc] = ld32cvt(arow, kc, kb0);
#pragma unroll
  for (int nt = 0; nt < NT; ++nt) {
    const _Float16* brow = W + (size_t)(nt * 16 + l15) * K;
    v8f acc = {};
#pragma unroll
    for (int kc = 0; kc < KC; ++kc) acc = wmma32(a[kc], ld16(brow, kc, kb0), acc);
    float bv = bias[nt * 16 + l15];
#pragma unroll
    for (int r = 0; r < 8; ++r) {
      float v = fmaxf(acc[r] + bv, 0.0f);
      outh[(size_t)(mbase + r + 8 * hi) * (NT * 16) + nt * 16 + l15] = (_Float16)v;
    }
  }
}

// ---------------------------------------------------------------- LSTM aggregator
// One wave owns a 16-node tile; h (A-layout, f16) lives in registers across all
// 16 timesteps; c (f32) lives in a wave-private LDS tile (elementwise use only,
// never a WMMA operand). h C->A relayout via wave-private LDS tile.
__global__ __launch_bounds__(128, 1)
void lstm_agg(const _Float16* __restrict__ hN, const _Float16* __restrict__ hE,
              const int* __restrict__ nidx, const int* __restrict__ eidx, // [N,16]
              const _Float16* __restrict__ Wih,  // [512,128] f16
              const _Float16* __restrict__ Whh,  // [512,128] f16
              const float* __restrict__ bsum,    // [512] = bih + bhh
              const _Float16* __restrict__ Wr,   // [64,128] f16
              const float* __restrict__ br,      // [64]
              _Float16* __restrict__ aggh,       // [N,64] f16 out
              int mtiles) {
  __shared__ __align__(16) _Float16 hTile[4][16][HH2];  // 16 KB
  __shared__ __align__(16) float    cTile[4][HH2][20];  // 40 KB (pad 20 dw/col)
  int wib = threadIdx.x >> 5;
  int wave = blockIdx.x * 4 + wib;
  if (wave >= mtiles) return;
  int lane = threadIdx.x & 31;
  int l15 = lane & 15, hi = lane >> 4, kb0 = hi * 8;
  int mbase = wave * 16;
  int node = mbase + l15;

  v16h hA[4];
#pragma unroll
  for (int kc = 0; kc < 4; ++kc)
#pragma unroll
    for (int i = 0; i < 16; ++i) hA[kc][i] = (_Float16)0.0f;
  // zero this wave's cell-state tile (each lane owns 8 floats per col-tile)
  {
    float4 z = {0.0f, 0.0f, 0.0f, 0.0f};
#pragma unroll
    for (int jt = 0; jt < 8; ++jt) {
      float4* cp = (float4*)&cTile[wib][jt * 16 + l15][hi * 8];
      cp[0] = z;
      cp[1] = z;
    }
  }

  for (int t = 0; t < DD; ++t) {
    // gather x_t = concat(hN[nidx[node,t]], hE[eidx[node,t]]) into A-layout
    int nr = nidx[node * DD + t];
    int er = eidx[node * DD + t];
    const _Float16* s0 = hN + (size_t)nr * 64;
    const _Float16* s1 = hE + (size_t)er * 64;
    v16h xA[4];
#pragma unroll
    for (int kc = 0; kc < 4; ++kc) {
      int k0 = kc * 32 + kb0, k1 = k0 + 16;
      uint4 u0 = *(const uint4*)(k0 < 64 ? s0 + k0 : s1 + (k0 - 64));
      uint4 u1 = *(const uint4*)(k1 < 64 ? s0 + k1 : s1 + (k1 - 64));
      xA[kc] = mk16(u0, u1);
    }
    // gates: col-tile jt covers cols [jt*16, jt*16+16) of each of i,f,g,o
#pragma unroll 1
    for (int jt = 0; jt < 8; ++jt) {
      v8f g[4];
#pragma unroll
      for (int gate = 0; gate < 4; ++gate) {
        int nt = gate * 8 + jt;
        const _Float16* bi = Wih + (size_t)(nt * 16 + l15) * HH2;
        const _Float16* bh = Whh + (size_t)(nt * 16 + l15) * HH2;
        v8f acc = {};
#pragma unroll
        for (int kc = 0; kc < 4; ++kc) acc = wmma32(xA[kc], ld16(bi, kc, kb0), acc);
#pragma unroll
        for (int kc = 0; kc < 4; ++kc) acc = wmma32(hA[kc], ld16(bh, kc, kb0), acc);
        float bv = bsum[nt * 16 + l15];
#pragma unroll
        for (int r = 0; r < 8; ++r) acc[r] += bv;
        g[gate] = acc;
      }
      // elementwise LSTM cell update; c lives in LDS (2x b128 in, 2x b128 out)
      float4* cp = (float4*)&cTile[wib][jt * 16 + l15][hi * 8];
      float4 c0 = cp[0], c1 = cp[1];
      float cv[8] = {c0.x, c0.y, c0.z, c0.w, c1.x, c1.y, c1.z, c1.w};
      float cn[8];
#pragma unroll
      for (int r = 0; r < 8; ++r) {
        float iv = sigm(g[0][r]);
        float fv = sigm(g[1][r]);
        float gv = tanh_f(g[2][r]);
        float ov = sigm(g[3][r]);
        float c2 = fv * cv[r] + iv * gv;
        cn[r] = c2;
        float hv = ov * tanh_f(c2);
        hTile[wib][r + 8 * hi][jt * 16 + l15] = (_Float16)hv;  // row-major h tile
      }
      float4 n0 = {cn[0], cn[1], cn[2], cn[3]};
      float4 n1 = {cn[4], cn[5], cn[6], cn[7]};
      cp[0] = n0;
      cp[1] = n1;
    }
    // reload h in A-layout for next step (wave-private; waitcnt handled by compiler)
    const _Float16* hrow = &hTile[wib][l15][0];
#pragma unroll
    for (int kc = 0; kc < 4; ++kc) hA[kc] = ld16(hrow, kc, kb0);
  }

  // agg = relu( relu(h_last) @ Wr^T + br )
#pragma unroll
  for (int kc = 0; kc < 4; ++kc)
#pragma unroll
    for (int i = 0; i < 16; ++i) {
      _Float16 v = hA[kc][i];
      hA[kc][i] = (float)v > 0.0f ? v : (_Float16)0.0f;
    }
#pragma unroll
  for (int nt = 0; nt < 4; ++nt) {
    const _Float16* bw = Wr + (size_t)(nt * 16 + l15) * HH2;
    v8f acc = {};
#pragma unroll
    for (int kc = 0; kc < 4; ++kc) acc = wmma32(hA[kc], ld16(bw, kc, kb0), acc);
    float bv = br[nt * 16 + l15];
#pragma unroll
    for (int r = 0; r < 8; ++r) {
      float v = fmaxf(acc[r] + bv, 0.0f);
      aggh[(size_t)(mbase + r + 8 * hi) * HH + nt * 16 + l15] = (_Float16)v;
    }
  }
}

// ---------------------------------------------------------------- final node GEMM
// node_out = relu( [agg_p | hN | agg_s](N,192) @ Wt_n^T + bt_n )  -> f32
__global__ __launch_bounds__(256)
void node_out_gemm(const _Float16* __restrict__ aggp, const _Float16* __restrict__ hN,
                   const _Float16* __restrict__ aggs, const _Float16* __restrict__ Wt,
                   const float* __restrict__ bt, float* __restrict__ out, int mtiles) {
  int gw = (int)((blockIdx.x * blockDim.x + threadIdx.x) >> 5);
  if (gw >= mtiles) return;
  int lane = threadIdx.x & 31;
  int l15 = lane & 15, hi = lane >> 4, kb0 = hi * 8;
  int mbase = gw * 16;
  int row = mbase + l15;
  const _Float16* s0 = aggp + (size_t)row * 64;
  const _Float16* s1 = hN + (size_t)row * 64;
  const _Float16* s2 = aggs + (size_t)row * 64;
  v16h a[6];
#pragma unroll
  for (int kc = 0; kc < 6; ++kc) {
    int k0 = kc * 32 + kb0, k1 = k0 + 16;
    a[kc] = mk16(*(const uint4*)segp(s0, s1, s2, k0),
                 *(const uint4*)segp(s0, s1, s2, k1));
  }
#pragma unroll
  for (int nt = 0; nt < 8; ++nt) {
    const _Float16* brow = Wt + (size_t)(nt * 16 + l15) * 192;
    v8f acc = {};
#pragma unroll
    for (int kc = 0; kc < 6; ++kc) acc = wmma32(a[kc], ld16(brow, kc, kb0), acc);
    float bv = bt[nt * 16 + l15];
#pragma unroll
    for (int r = 0; r < 8; ++r)
      out[(size_t)(mbase + r + 8 * hi) * OUTN + nt * 16 + l15] =
          fmaxf(acc[r] + bv, 0.0f);
  }
}

// ---------------------------------------------------------------- final edge GEMM
// edge_out = relu( [hN[src] | hE | hN[dst]](E,192) @ Wt_e^T + bt_e ) -> f32
__global__ __launch_bounds__(256)
void edge_out_gemm(const _Float16* __restrict__ hN, const _Float16* __restrict__ hE,
                   const int* __restrict__ src, const int* __restrict__ dst,
                   const _Float16* __restrict__ Wt, const float* __restrict__ bt,
                   float* __restrict__ out, int mtiles) {
  int gw = (int)((blockIdx.x * blockDim.x + threadIdx.x) >> 5);
  if (gw >= mtiles) return;
  int lane = threadIdx.x & 31;
  int l15 = lane & 15, hi = lane >> 4, kb0 = hi * 8;
  int mbase = gw * 16;
  int row = mbase + l15;
  int sr = src[row], dr = dst[row];
  const _Float16* s0 = hN + (size_t)sr * 64;
  const _Float16* s1 = hE + (size_t)row * 64;
  const _Float16* s2 = hN + (size_t)dr * 64;
  v16h a[6];
#pragma unroll
  for (int kc = 0; kc < 6; ++kc) {
    int k0 = kc * 32 + kb0, k1 = k0 + 16;
    a[kc] = mk16(*(const uint4*)segp(s0, s1, s2, k0),
                 *(const uint4*)segp(s0, s1, s2, k1));
  }
#pragma unroll
  for (int nt = 0; nt < 4; ++nt) {
    const _Float16* brow = Wt + (size_t)(nt * 16 + l15) * 192;
    v8f acc = {};
#pragma unroll
    for (int kc = 0; kc < 6; ++kc) acc = wmma32(a[kc], ld16(brow, kc, kb0), acc);
    float bv = bt[nt * 16 + l15];
#pragma unroll
    for (int r = 0; r < 8; ++r)
      out[(size_t)(mbase + r + 8 * hi) * OUTE + nt * 16 + l15] =
          fmaxf(acc[r] + bv, 0.0f);
  }
}

// ---------------------------------------------------------------- host

extern "C" void kernel_launch(void* const* d_in, const int* in_sizes, int n_in,
                              void* d_out, int out_size, void* d_ws, size_t ws_size,
                              hipStream_t stream) {
  (void)in_sizes; (void)n_in; (void)out_size; (void)ws_size;
  const float* node_vecs = (const float*)d_in[0];
  const float* edge_vecs = (const float*)d_in[1];
  const int* pni  = (const int*)d_in[2];
  const int* pei  = (const int*)d_in[3];
  const int* sni  = (const int*)d_in[4];
  const int* sei  = (const int*)d_in[5];
  const int* esrc = (const int*)d_in[6];
  const int* edst = (const int*)d_in[7];
  const float* Wn    = (const float*)d_in[8];
  const float* bn    = (const float*)d_in[9];
  const float* We    = (const float*)d_in[10];
  const float* be    = (const float*)d_in[11];
  const float* Wih_p = (const float*)d_in[12];
  const float* Whh_p = (const float*)d_in[13];
  const float* bih_p = (const float*)d_in[14];
  const float* bhh_p = (const float*)d_in[15];
  const float* Wr_p  = (const float*)d_in[16];
  const float* br_p  = (const float*)d_in[17];
  const float* Wih_s = (const float*)d_in[18];
  const float* Whh_s = (const float*)d_in[19];
  const float* bih_s = (const float*)d_in[20];
  const float* bhh_s = (const float*)d_in[21];
  const float* Wr_s  = (const float*)d_in[22];
  const float* br_s  = (const float*)d_in[23];
  const float* Wt_n  = (const float*)d_in[24];
  const float* bt_n  = (const float*)d_in[25];
  const float* Wt_e  = (const float*)d_in[26];
  const float* bt_e  = (const float*)d_in[27];

  char* ws = (char*)d_ws;
  size_t off = 0;
  auto alloc = [&](size_t bytes) -> char* {
    char* p = ws + off;
    off += (bytes + 255) & ~(size_t)255;
    return p;
  };
  _Float16* Wn_h    = (_Float16*)alloc((size_t)HH * INN * 2);
  _Float16* We_h    = (_Float16*)alloc((size_t)HH * INE * 2);
  _Float16* Wih_p_h = (_Float16*)alloc((size_t)G4 * HH2 * 2);
  _Float16* Whh_p_h = (_Float16*)alloc((size_t)G4 * HH2 * 2);
  _Float16* Wr_p_h  = (_Float16*)alloc((size_t)HH * HH2 * 2);
  _Float16* Wih_s_h = (_Float16*)alloc((size_t)G4 * HH2 * 2);
  _Float16* Whh_s_h = (_Float16*)alloc((size_t)G4 * HH2 * 2);
  _Float16* Wr_s_h  = (_Float16*)alloc((size_t)HH * HH2 * 2);
  _Float16* Wt_n_h  = (_Float16*)alloc((size_t)OUTN * 192 * 2);
  _Float16* Wt_e_h  = (_Float16*)alloc((size_t)OUTE * 192 * 2);
  float*    bsum_p  = (float*)alloc((size_t)G4 * 4);
  float*    bsum_s  = (float*)alloc((size_t)G4 * 4);
  _Float16* hN_h    = (_Float16*)alloc((size_t)NN * HH * 2);
  _Float16* hE_h    = (_Float16*)alloc((size_t)EE * HH * 2);
  _Float16* aggp_h  = (_Float16*)alloc((size_t)NN * HH * 2);
  _Float16* aggs_h  = (_Float16*)alloc((size_t)NN * HH * 2);

  auto cvt = [&](const float* s, _Float16* d, int n) {
    cvt_f32_to_f16<<<(n + 255) / 256, 256, 0, stream>>>(s, d, n);
  };
  cvt(Wn, Wn_h, HH * INN);
  cvt(We, We_h, HH * INE);
  cvt(Wih_p, Wih_p_h, G4 * HH2);
  cvt(Whh_p, Whh_p_h, G4 * HH2);
  cvt(Wr_p, Wr_p_h, HH * HH2);
  cvt(Wih_s, Wih_s_h, G4 * HH2);
  cvt(Whh_s, Whh_s_h, G4 * HH2);
  cvt(Wr_s, Wr_s_h, HH * HH2);
  cvt(Wt_n, Wt_n_h, OUTN * 192);
  cvt(Wt_e, Wt_e_h, OUTE * 192);
  add_bias<<<(G4 + 255) / 256, 256, 0, stream>>>(bih_p, bhh_p, bsum_p, G4);
  add_bias<<<(G4 + 255) / 256, 256, 0, stream>>>(bih_s, bhh_s, bsum_s, G4);

  const int mtN = NN / 16;  // 3125
  const int mtE = EE / 16;  // 50000
  // hN = relu(node_vecs @ Wn^T + bn), hE = relu(edge_vecs @ We^T + be)
  pretrans_gemm<4, 4><<<(mtN + 7) / 8, 256, 0, stream>>>(node_vecs, Wn_h, bn, hN_h, mtN);
  pretrans_gemm<2, 4><<<(mtE + 7) / 8, 256, 0, stream>>>(edge_vecs, We_h, be, hE_h, mtE);

  // LSTM aggregators (whole 16-step recurrence inside one launch each)
  lstm_agg<<<(mtN + 3) / 4, 128, 0, stream>>>(hN_h, hE_h, pni, pei, Wih_p_h, Whh_p_h,
                                              bsum_p, Wr_p_h, br_p, aggp_h, mtN);
  lstm_agg<<<(mtN + 3) / 4, 128, 0, stream>>>(hN_h, hE_h, sni, sei, Wih_s_h, Whh_s_h,
                                              bsum_s, Wr_s_h, br_s, aggs_h, mtN);

  float* node_out = (float*)d_out;
  float* edge_out = node_out + (size_t)NN * OUTN;
  node_out_gemm<<<(mtN + 7) / 8, 256, 0, stream>>>(aggp_h, hN_h, aggs_h, Wt_n_h, bt_n,
                                                   node_out, mtN);
  edge_out_gemm<<<(mtE + 7) / 8, 256, 0, stream>>>(hN_h, hE_h, esrc, edst, Wt_e_h, bt_e,
                                                   edge_out, mtE);
}
